// MixModel_57930518889106
// MI455X (gfx1250) — matrix operand order repaired and checked by reading, not compiled, not run on hardware
//
#include <hip/hip_runtime.h>

typedef __attribute__((ext_vector_type(16))) _Float16 v16h;
typedef __attribute__((ext_vector_type(8)))  _Float16 v8h_t;
typedef __attribute__((ext_vector_type(8)))  float    v8f;

#define NNODES 100000
#define NEDGES 1600000
#define NFEAT  512
#define FDIM   128
#define NCLS   10

#define CDIV(a,b) (((a)+(b)-1)/(b))

// ---------------- conversion kernels ----------------

__global__ void k_f32_to_f16(const float* __restrict__ in, _Float16* __restrict__ out, long n) {
  long i = (long)blockIdx.x * blockDim.x + threadIdx.x;
  if (i < n) out[i] = (_Float16)in[i];
}

// Wt[n][k] = W[k][n];  W row-major [K][N] f32 -> Wt row-major [N][K] f16
__global__ void k_weight_transpose(const float* __restrict__ W, _Float16* __restrict__ Wt, int K, int N) {
  int i = blockIdx.x * blockDim.x + threadIdx.x;
  if (i >= K * N) return;
  int n = i / K, k = i % K;
  Wt[i] = (_Float16)W[k * N + n];
}

// classifier weight [128][10] -> padded transposed [16][128], rows >= 10 zero
__global__ void k_wc_transpose(const float* __restrict__ W, _Float16* __restrict__ Wt) {
  int i = blockIdx.x * blockDim.x + threadIdx.x;
  if (i >= 16 * FDIM) return;
  int n = i / FDIM, k = i % FDIM;
  Wt[i] = (n < NCLS) ? (_Float16)W[k * NCLS + n] : (_Float16)0.f;
}

// ---------------- degree kernels ----------------

__global__ void k_deg_count(const int* __restrict__ dst, float* __restrict__ deg, int E) {
  int i = blockIdx.x * blockDim.x + threadIdx.x;
  if (i < E) atomicAdd(&deg[dst[i]], 1.0f);
}

// in: raw count -> out: dis = rsqrt(deg+1); invd = 1/(deg+1)
__global__ void k_deg_finalize(float* __restrict__ dis, float* __restrict__ invd, int n) {
  int i = blockIdx.x * blockDim.x + threadIdx.x;
  if (i >= n) return;
  float d = dis[i] + 1.0f;
  dis[i]  = rsqrtf(d);
  invd[i] = 1.0f / d;
}

// ---------------- WMMA GEMM:  C[M x NT*16] = A[M x K](f16) * Bt^T  (+bias, relu) ----------
// Bt stored transposed: Bt[N][K] row-major f16.
template<int NT>
__global__ void __launch_bounds__(128)
k_gemm_wmma(const _Float16* __restrict__ A, const _Float16* __restrict__ Bt,
            const float* __restrict__ bias, float* __restrict__ Cf,
            _Float16* __restrict__ Ch, int M, int K, int relu) {
  const int lane  = threadIdx.x & 31;
  const int wave  = threadIdx.x >> 5;
  const int m0    = blockIdx.x * 64 + wave * 16;
  if (m0 >= M) return;                       // whole wave exits (M % 16 == 0)
  const int lmod  = lane & 15;
  const int khalf = lane >> 4;               // 0 or 1
  const int mrow  = m0 + lmod;

  v8f acc[NT] = {};

  const _Float16* arow = A + (long)mrow * K;
  for (int k = 0; k < K; k += 32) {
    const int kb = k + khalf * 16;
    v8h_t alo = *(const v8h_t*)(arow + kb);
    v8h_t ahi = *(const v8h_t*)(arow + kb + 8);
    v16h a = __builtin_shufflevector(alo, ahi, 0,1,2,3,4,5,6,7,8,9,10,11,12,13,14,15);
    const int kk = k + khalf * 16;
#pragma unroll
    for (int t = 0; t < NT; ++t) {
      v16h b = *(const v16h*)(Bt + (long)(t * 16 + lmod) * K + kk);
      acc[t] = __builtin_amdgcn_wmma_f32_16x16x32_f16(
          false, a, false, b, (short)0, acc[t], false, false);
    }
  }

  const int N = NT * 16;
#pragma unroll
  for (int t = 0; t < NT; ++t) {
    float bv = bias ? bias[t * 16 + lmod] : 0.f;
#pragma unroll
    for (int r = 0; r < 8; ++r) {
      int row = m0 + r + 8 * khalf;
      float v = acc[t][r] + bv;
      if (relu) v = fmaxf(v, 0.f);
      long off = (long)row * N + t * 16 + lmod;
      if (Cf) Cf[off] = v;
      if (Ch) Ch[off] = (_Float16)v;
    }
  }
}

// ---------------- edge scatter-aggregation ----------------
// agg[dst[e]] += h[(perm?perm:id)(src[e])] * dis[src[e]]*dis[dst[e]]   (D floats per edge)
__global__ void k_agg_scatter(const float* __restrict__ h, const int* __restrict__ src,
                              const int* __restrict__ dst, const int* __restrict__ perm,
                              const float* __restrict__ dis, float* __restrict__ agg,
                              int E, int D) {
  long tid = (long)blockIdx.x * blockDim.x + threadIdx.x;
  const int qpd = D >> 2;
  if (tid >= (long)E * qpd) return;
  int e = (int)(tid / qpd);
  int q = (int)(tid % qpd);
  int s = src[e], d = dst[e];
  float w = dis[s] * dis[d];
  int sg = perm ? perm[s] : s;
  const float4 v = *(const float4*)(h + (long)sg * D + q * 4);
  float* ap = agg + (long)d * D + q * 4;
  atomicAdd(ap + 0, v.x * w);
  atomicAdd(ap + 1, v.y * w);
  atomicAdd(ap + 2, v.z * w);
  atomicAdd(ap + 3, v.w * w);
}

// ---------------- GCN epilogue: out = agg + h[(perm?)i]*invdeg[i] + bias  (opt relu) -------
__global__ void k_gcn_epilogue(const float* __restrict__ agg, const float* __restrict__ h,
                               const int* __restrict__ perm, const float* __restrict__ invd,
                               const float* __restrict__ bias,
                               float* __restrict__ of32, int ld32,
                               _Float16* __restrict__ of16,
                               int n, int D, int DS, int relu) {
  long tid = (long)blockIdx.x * blockDim.x + threadIdx.x;
  if (tid >= (long)n * DS) return;
  int i = (int)(tid / DS), f = (int)(tid % DS);
  int ig = perm ? perm[i] : i;
  float v = agg[(long)i * D + f] + h[(long)ig * D + f] * invd[i] + bias[f];
  if (relu) v = fmaxf(v, 0.f);
  if (of32) of32[(long)i * ld32 + f] = v;
  if (of16) of16[(long)i * D + f] = (_Float16)v;
}

// ---------------- per-node bilinear score: sigmoid(dot(e3[n], U[n])), wave per node --------
__global__ void k_score(const float* __restrict__ e3, const float* __restrict__ U,
                        float* __restrict__ out, int n, int col) {
  long gid = (long)blockIdx.x * blockDim.x + threadIdx.x;
  int node = (int)(gid >> 5);
  int lane = (int)(gid & 31);
  if (node >= n) return;
  const float4 a = *(const float4*)(e3 + (long)node * FDIM + lane * 4);
  const float4 b = *(const float4*)(U  + (long)node * FDIM + lane * 4);
  float s = a.x * b.x + a.y * b.y + a.z * b.z + a.w * b.w;
#pragma unroll
  for (int m = 16; m >= 1; m >>= 1) s += __shfl_xor(s, m, 32);
  if (lane == 0) out[(long)node * 12 + col] = 1.f / (1.f + __expf(-s));
}

// ---------------- driver ----------------

extern "C" void kernel_launch(void* const* d_in, const int* in_sizes, int n_in,
                              void* d_out, int out_size, void* d_ws, size_t ws_size,
                              hipStream_t stream) {
  (void)in_sizes; (void)n_in; (void)out_size; (void)ws_size;

  const float* x   = (const float*)d_in[0];
  const int*   ei  = (const int*)d_in[1];
  const int*   eih = (const int*)d_in[2];
  const int*   perm= (const int*)d_in[4];
  const float* W1  = (const float*)d_in[5];
  const float* b1  = (const float*)d_in[6];
  const float* W2  = (const float*)d_in[7];
  const float* b2  = (const float*)d_in[8];
  const float* W3  = (const float*)d_in[9];
  const float* b3  = (const float*)d_in[10];
  const float* M1  = (const float*)d_in[11];
  const float* mb1 = (const float*)d_in[12];
  const float* M2  = (const float*)d_in[13];
  const float* mb2 = (const float*)d_in[14];
  const float* Wc  = (const float*)d_in[15];
  const float* bc  = (const float*)d_in[16];
  const float* Wd  = (const float*)d_in[17];
  float* out = (float*)d_out;

  const int* src  = ei;        const int* dst  = ei  + NEDGES;
  const int* hsrc = eih;       const int* hdst = eih + NEDGES;

  // ---- bump allocator in workspace ----
  char* p = (char*)d_ws;
  auto alloc = [&](size_t bytes) -> void* {
    void* r = (void*)p;
    p += (bytes + 255) & ~(size_t)255;
    return r;
  };
  _Float16* xh   = (_Float16*)alloc((size_t)NNODES * NFEAT * 2);
  _Float16* w1t  = (_Float16*)alloc((size_t)NFEAT * FDIM * 2);
  _Float16* w2t  = (_Float16*)alloc((size_t)FDIM * FDIM * 2);
  _Float16* w3t  = (_Float16*)alloc((size_t)FDIM * FDIM * 2);
  _Float16* m1t  = (_Float16*)alloc((size_t)FDIM * FDIM * 2);
  _Float16* m2t  = (_Float16*)alloc((size_t)FDIM * FDIM * 2);
  _Float16* wdt  = (_Float16*)alloc((size_t)FDIM * FDIM * 2);  // Bt for embed2 @ Wd^T is Wd itself
  _Float16* wct  = (_Float16*)alloc((size_t)16 * FDIM * 2);
  float* dis   = (float*)alloc((size_t)NNODES * 4);
  float* invd  = (float*)alloc((size_t)NNODES * 4);
  float* dish  = (float*)alloc((size_t)NNODES * 4);
  float* invdh = (float*)alloc((size_t)NNODES * 4);
  const size_t FB = (size_t)NNODES * FDIM * 4;   // f32 feature buffer bytes
  float* hbuf   = (float*)alloc(FB);             // GEMM f32 outputs / embed3 / reuse
  float* aggbuf = (float*)alloc(FB);             // scatter target / U / reuse
  _Float16* e_l1  = (_Float16*)alloc(FB / 2);    // layer1 normal -> later embed2
  _Float16* e_l1b = (_Float16*)alloc(FB / 2);    // layer1 bad    -> later embed2_bad
  _Float16* e1    = (_Float16*)alloc(FB / 2);    // embed1
  _Float16* e1b   = (_Float16*)alloc(FB / 2);    // embed1_bad    -> later MLP hidden t
  float* hc   = (float*)alloc((size_t)NNODES * 16 * 4);
  float* aggc = (float*)alloc((size_t)NNODES * 16 * 4);

  // ---- one-time conversions ----
  k_f32_to_f16<<<(int)CDIV((long)NNODES * NFEAT, 256), 256, 0, stream>>>(x, xh, (long)NNODES * NFEAT);
  k_weight_transpose<<<CDIV(NFEAT * FDIM, 256), 256, 0, stream>>>(W1, w1t, NFEAT, FDIM);
  k_weight_transpose<<<CDIV(FDIM * FDIM, 256), 256, 0, stream>>>(W2, w2t, FDIM, FDIM);
  k_weight_transpose<<<CDIV(FDIM * FDIM, 256), 256, 0, stream>>>(W3, w3t, FDIM, FDIM);
  k_weight_transpose<<<CDIV(FDIM * FDIM, 256), 256, 0, stream>>>(M1, m1t, FDIM, FDIM);
  k_weight_transpose<<<CDIV(FDIM * FDIM, 256), 256, 0, stream>>>(M2, m2t, FDIM, FDIM);
  k_f32_to_f16<<<CDIV(FDIM * FDIM, 256), 256, 0, stream>>>(Wd, wdt, (long)FDIM * FDIM);
  k_wc_transpose<<<CDIV(16 * FDIM, 256), 256, 0, stream>>>(Wc, wct);

  // ---- degrees ----
  hipMemsetAsync(dis,  0, (size_t)NNODES * 4, stream);
  hipMemsetAsync(dish, 0, (size_t)NNODES * 4, stream);
  k_deg_count<<<CDIV(NEDGES, 256), 256, 0, stream>>>(dst,  dis,  NEDGES);
  k_deg_count<<<CDIV(NEDGES, 256), 256, 0, stream>>>(hdst, dish, NEDGES);
  k_deg_finalize<<<CDIV(NNODES, 256), 256, 0, stream>>>(dis,  invd,  NNODES);
  k_deg_finalize<<<CDIV(NNODES, 256), 256, 0, stream>>>(dish, invdh, NNODES);

  const int GB = CDIV(NNODES, 64);
  const int EG = (int)CDIV((long)NEDGES * (FDIM / 4), 256);
  const int NG = (int)CDIV((long)NNODES * FDIM, 256);

  // 1. h1 = x @ W1
  k_gemm_wmma<8><<<GB, 128, 0, stream>>>(xh, w1t, (const float*)nullptr, hbuf, (_Float16*)nullptr, NNODES, NFEAT, 0);
  // 2. layer1 normal: aggregate -> relu -> e_l1 (f16)
  hipMemsetAsync(aggbuf, 0, FB, stream);
  k_agg_scatter<<<EG, 256, 0, stream>>>(hbuf, src, dst, (const int*)nullptr, dis, aggbuf, NEDGES, FDIM);
  k_gcn_epilogue<<<NG, 256, 0, stream>>>(aggbuf, hbuf, (const int*)nullptr, invd, b1, (float*)nullptr, 0, e_l1, NNODES, FDIM, FDIM, 1);
  // 3. layer1 bad: x[perm]@W1 == h1[perm] gathered via perm
  hipMemsetAsync(aggbuf, 0, FB, stream);
  k_agg_scatter<<<EG, 256, 0, stream>>>(hbuf, src, dst, perm, dis, aggbuf, NEDGES, FDIM);
  k_gcn_epilogue<<<NG, 256, 0, stream>>>(aggbuf, hbuf, perm, invd, b1, (float*)nullptr, 0, e_l1b, NNODES, FDIM, FDIM, 1);
  // 4-5. layer2 normal -> e1
  k_gemm_wmma<8><<<GB, 128, 0, stream>>>(e_l1, w2t, (const float*)nullptr, hbuf, (_Float16*)nullptr, NNODES, FDIM, 0);
  hipMemsetAsync(aggbuf, 0, FB, stream);
  k_agg_scatter<<<EG, 256, 0, stream>>>(hbuf, src, dst, (const int*)nullptr, dis, aggbuf, NEDGES, FDIM);
  k_gcn_epilogue<<<NG, 256, 0, stream>>>(aggbuf, hbuf, (const int*)nullptr, invd, b2, (float*)nullptr, 0, e1, NNODES, FDIM, FDIM, 1);
  // 6. layer2 bad -> e1b
  k_gemm_wmma<8><<<GB, 128, 0, stream>>>(e_l1b, w2t, (const float*)nullptr, hbuf, (_Float16*)nullptr, NNODES, FDIM, 0);
  hipMemsetAsync(aggbuf, 0, FB, stream);
  k_agg_scatter<<<EG, 256, 0, stream>>>(hbuf, src, dst, (const int*)nullptr, dis, aggbuf, NEDGES, FDIM);
  k_gcn_epilogue<<<NG, 256, 0, stream>>>(aggbuf, hbuf, (const int*)nullptr, invd, b2, (float*)nullptr, 0, e1b, NNODES, FDIM, FDIM, 1);
  // 7. encoder2 normal (hop edges, no relu) -> embed2 stored in e_l1 (reuse)
  k_gemm_wmma<8><<<GB, 128, 0, stream>>>(e1, w3t, (const float*)nullptr, hbuf, (_Float16*)nullptr, NNODES, FDIM, 0);
  hipMemsetAsync(aggbuf, 0, FB, stream);
  k_agg_scatter<<<EG, 256, 0, stream>>>(hbuf, hsrc, hdst, (const int*)nullptr, dish, aggbuf, NEDGES, FDIM);
  k_gcn_epilogue<<<NG, 256, 0, stream>>>(aggbuf, hbuf, (const int*)nullptr, invdh, b3, (float*)nullptr, 0, e_l1, NNODES, FDIM, FDIM, 0);
  // 8. encoder2 bad -> embed2_bad in e_l1b (reuse)
  k_gemm_wmma<8><<<GB, 128, 0, stream>>>(e1b, w3t, (const float*)nullptr, hbuf, (_Float16*)nullptr, NNODES, FDIM, 0);
  hipMemsetAsync(aggbuf, 0, FB, stream);
  k_agg_scatter<<<EG, 256, 0, stream>>>(hbuf, hsrc, hdst, (const int*)nullptr, dish, aggbuf, NEDGES, FDIM);
  k_gcn_epilogue<<<NG, 256, 0, stream>>>(aggbuf, hbuf, (const int*)nullptr, invdh, b3, (float*)nullptr, 0, e_l1b, NNODES, FDIM, FDIM, 0);
  // 9. MLP hidden: t = relu(embed1 @ M1 + mb1) -> f16 in e1b (reuse, embed1_bad dead)
  k_gemm_wmma<8><<<GB, 128, 0, stream>>>(e1, m1t, mb1, (float*)nullptr, e1b, NNODES, FDIM, 1);
  // 10. embed3 = t @ M2 + mb2 -> f32 in hbuf
  k_gemm_wmma<8><<<GB, 128, 0, stream>>>(e1b, m2t, mb2, hbuf, (_Float16*)nullptr, NNODES, FDIM, 0);
  // 11. U = embed2 @ Wd^T -> aggbuf; score = sigmoid(<embed3, U>) -> out col 10
  k_gemm_wmma<8><<<GB, 128, 0, stream>>>(e_l1, wdt, (const float*)nullptr, aggbuf, (_Float16*)nullptr, NNODES, FDIM, 0);
  k_score<<<(int)CDIV((long)NNODES * 32, 256), 256, 0, stream>>>(hbuf, aggbuf, out, NNODES, 10);
  // 12. U_bad; score_bad -> out col 11
  k_gemm_wmma<8><<<GB, 128, 0, stream>>>(e_l1b, wdt, (const float*)nullptr, aggbuf, (_Float16*)nullptr, NNODES, FDIM, 0);
  k_score<<<(int)CDIV((long)NNODES * 32, 256), 256, 0, stream>>>(hbuf, aggbuf, out, NNODES, 11);
  // 13. classifier GCN: hc = embed1 @ Wc (padded to 16 cols) -> aggregate -> out cols 0..9
  k_gemm_wmma<1><<<GB, 128, 0, stream>>>(e1, wct, (const float*)nullptr, hc, (_Float16*)nullptr, NNODES, FDIM, 0);
  hipMemsetAsync(aggc, 0, (size_t)NNODES * 16 * 4, stream);
  k_agg_scatter<<<(int)CDIV((long)NEDGES * 4, 256), 256, 0, stream>>>(hc, src, dst, (const int*)nullptr, dis, aggc, NEDGES, 16);
  k_gcn_epilogue<<<(int)CDIV((long)NNODES * NCLS, 256), 256, 0, stream>>>(aggc, hc, (const int*)nullptr, invd, bc, out, 12, (_Float16*)nullptr, NNODES, 16, NCLS, 0);
}
